// NNUEMCTSModel_58291296141587
// MI455X (gfx1250) — matrix operand order repaired and checked by reading, not compiled, not run on hardware
//
#include <hip/hip_runtime.h>

#define FT 256          // feature transform width
#define KP 608          // fc1 input dim padded 578 -> 608 (multiple of 32)
#define ROWSTRIDE 616   // bf16 elems per LDS activation row (1232 B: 16B aligned, bank-staggered)
#define ROWS_PER_WG 64
#define HID 256         // hidden dim
#define HSTRIDE 264     // bf16 elems per LDS h row (528 B: 16B aligned, bank-staggered)
#define N2 64           // fc2 fused output cols: 60 policy + 1 value + 3 pad

typedef __attribute__((ext_vector_type(16))) __bf16 v16bf;
typedef __attribute__((ext_vector_type(8)))  __bf16 v8bf;
typedef __attribute__((ext_vector_type(8)))  float  v8f;

// ---------------------------------------------------------------------------
// Prep: fc1 weights -> bf16 transposed [N=256][K=608]; fc2p + fc2v fused
// -> bf16 [N=64][K=256] (col 60 = value head), both into d_ws.
// ---------------------------------------------------------------------------
__global__ __launch_bounds__(256) void nnue_convert_weights(
    const float* __restrict__ fc1_w,   // [578][256]
    const float* __restrict__ fc2p_w,  // [256][60]
    const float* __restrict__ fc2v_w,  // [256]
    unsigned short* __restrict__ ws)
{
  __bf16* wT1 = (__bf16*)ws;             // [256][KP]
  __bf16* wT2 = (__bf16*)ws + 256 * KP;  // [N2][HID]
  const int i = blockIdx.x * 256 + threadIdx.x;
  const int n1 = 256 * KP;
  if (i < n1) {
    const int n = i / KP, k = i % KP;
    const float v = (k < 578) ? fc1_w[k * 256 + n] : 0.f;
    wT1[i] = (__bf16)v;
  } else {
    const int j = i - n1;
    if (j < N2 * HID) {
      const int n = j / HID, k = j % HID;
      const float v = (n < 60) ? fc2p_w[k * 60 + n]
                               : ((n == 60) ? fc2v_w[k] : 0.f);
      wT2[j] = (__bf16)v;
    }
  }
}

// A fragment for v_wmma_f32_16x16x32_bf16:
// lane<16 holds K {0..15}, lane>=16 holds K {16..31} of row (lane&15).
__device__ __forceinline__ v16bf load_afrag(const __bf16* rowp, int k0, int khalf) {
  const v8bf a0 = *(const v8bf*)(rowp + k0 + khalf * 16);
  const v8bf a1 = *(const v8bf*)(rowp + k0 + khalf * 16 + 8);
  return __builtin_shufflevector(a0, a1,
      0, 1, 2, 3, 4, 5, 6, 7, 8, 9, 10, 11, 12, 13, 14, 15);
}

// ---------------------------------------------------------------------------
// Fused NNUE forward. 256 threads = 8 wave32. Wave w: row-tile rt = w>>1
// (16 rows), column half ch = w&1 (128 of 256 hidden cols). 64 rows per WG.
// ---------------------------------------------------------------------------
__global__ __launch_bounds__(256) void nnue_fused_kernel(
    const int*   __restrict__ stm_idx,   // [B][3]
    const int*   __restrict__ nstm_idx,  // [B][3]
    const float* __restrict__ dense,     // [B][66]
    const float* __restrict__ ft_w,      // [120][256]
    const float* __restrict__ ft_b,      // [256]
    const float* __restrict__ fc1_b,     // [256]
    const float* __restrict__ fc2p_b,    // [60]
    const float* __restrict__ fc2v_b,    // [1]
    const unsigned short* __restrict__ ws,
    float* __restrict__ out_policy,      // [B][60]
    float* __restrict__ out_value,       // [B]
    int Btot)
{
  extern __shared__ char smem_raw[];
  __bf16* acts = (__bf16*)smem_raw;                                  // [64][ROWSTRIDE]
  __bf16* hbuf = (__bf16*)(smem_raw + ROWS_PER_WG * ROWSTRIDE * 2);  // [64][HSTRIDE]
  const __bf16* wT1 = (const __bf16*)ws;
  const __bf16* wT2 = (const __bf16*)ws + 256 * KP;

  const int tid  = threadIdx.x;
  const int lane = tid & 31;
  const int wave = tid >> 5;
  const int wgBase = blockIdx.x * ROWS_PER_WG;

  // ---------------- Phase A: bf16 activation panel (8 rows per wave) ------
  {
    const int f0 = lane * 8;  // 32 lanes x 8 features = 256
    for (int rr = 0; rr < 8; ++rr) {
      const int r  = wave * 8 + rr;
      const int rg = wgBase + r;
      __bf16* row = acts + r * ROWSTRIDE;
#pragma unroll
      for (int side = 0; side < 2; ++side) {
        const int* ip = (side ? nstm_idx : stm_idx) + rg * 3;
        float s[8] = {0.f, 0.f, 0.f, 0.f, 0.f, 0.f, 0.f, 0.f};
        if (rg < Btot) {
          const int i0 = ip[0], i1 = ip[1], i2 = ip[2];  // scalar loads
          const float4* p0 = (const float4*)(ft_w + i0 * FT + f0);
          const float4* p1 = (const float4*)(ft_w + i1 * FT + f0);
          const float4* p2 = (const float4*)(ft_w + i2 * FT + f0);
          const float4* pb = (const float4*)(ft_b + f0);
#pragma unroll
          for (int h = 0; h < 2; ++h) {
            const float4 a = p0[h], b = p1[h], c = p2[h], d = pb[h];
            s[h*4+0] = fmaxf(a.x + b.x + c.x + d.x, 0.f);
            s[h*4+1] = fmaxf(a.y + b.y + c.y + d.y, 0.f);
            s[h*4+2] = fmaxf(a.z + b.z + c.z + d.z, 0.f);
            s[h*4+3] = fmaxf(a.w + b.w + c.w + d.w, 0.f);
          }
        }
        v8bf pk;
#pragma unroll
        for (int e = 0; e < 8; ++e) pk[e] = (__bf16)s[e];
        *(v8bf*)(row + side * FT + f0) = pk;   // 16B LDS store, conflict-free
      }
#pragma unroll
      for (int t = 0; t < 3; ++t) {            // dense 66 + zero pad to 608
        const int f = lane + t * 32;           // 0..95
        const float v = (f < 66 && rg < Btot) ? dense[rg * 66 + f] : 0.f;
        row[2 * FT + f] = (__bf16)v;
      }
    }
  }
  __syncthreads();

  const int c16   = lane & 15;
  const int khalf = lane >> 4;
  const int rt    = wave >> 1;   // row tile 0..3
  const int ch    = wave & 1;    // column half 0..1
  const __bf16* myrow = acts + (rt * 16 + c16) * ROWSTRIDE;
  const v8f vzero = {0.f, 0.f, 0.f, 0.f, 0.f, 0.f, 0.f, 0.f};

  // ---------------- fc1: [16 x 608] @ [608 x 128] per wave ----------------
  // nt outer / K inner: B loads are independent of the accumulate chain, so
  // they stream ahead of the wmmas; only 1 live accumulator per chain.
  v8f acc[8];
#pragma unroll
  for (int nt = 0; nt < 8; ++nt) {
    v8f a = vzero;
    const __bf16* bcol = wT1 + (ch * 128 + nt * 16 + c16) * KP + khalf * 16;
#pragma unroll
    for (int ks = 0; ks < KP / 32; ++ks) {
      const v16bf af = load_afrag(myrow, ks * 32, khalf);
      const v16bf bf = *(const v16bf*)(bcol + ks * 32);
      a = __builtin_amdgcn_wmma_f32_16x16x32_bf16(
          false, af, false, bf, (short)0, a, false, false);
    }
    acc[nt] = a;
  }

  // Epilogue 1: bias + ReLU -> h panel (separate LDS buffer, no aliasing)
#pragma unroll
  for (int nt = 0; nt < 8; ++nt) {
    const int col = ch * 128 + nt * 16 + c16;
    const float b = fc1_b[col];
#pragma unroll
    for (int j = 0; j < 8; ++j) {
      const int m = khalf * 8 + j;  // C layout: lane<16 -> M=j, lane>=16 -> M=8+j
      hbuf[(rt * 16 + m) * HSTRIDE + col] = (__bf16)fmaxf(acc[nt][j] + b, 0.f);
    }
  }
  __syncthreads();  // wave pair wrote complementary column halves of h

  // ---------------- fc2 (policy+value fused): [16 x 256] @ [256 x 32] -----
  const __bf16* hrow = hbuf + (rt * 16 + c16) * HSTRIDE;
  v8f acc2[2];
#pragma unroll
  for (int nt = 0; nt < 2; ++nt) {
    v8f a = vzero;
    const __bf16* bcol = wT2 + ((ch * 2 + nt) * 16 + c16) * HID + khalf * 16;
#pragma unroll
    for (int ks = 0; ks < HID / 32; ++ks) {
      const v16bf af = load_afrag(hrow, ks * 32, khalf);
      const v16bf bf = *(const v16bf*)(bcol + ks * 32);
      a = __builtin_amdgcn_wmma_f32_16x16x32_bf16(
          false, af, false, bf, (short)0, a, false, false);
    }
    acc2[nt] = a;
  }

  // Epilogue 2: policy bias add; value = tanh(col 60 + bias)
  const float vbias = fc2v_b[0];
#pragma unroll
  for (int nt = 0; nt < 2; ++nt) {
    const int col = (ch * 2 + nt) * 16 + c16;
    const float pb = (col < 60) ? fc2p_b[col] : 0.f;
#pragma unroll
    for (int j = 0; j < 8; ++j) {
      const int m  = khalf * 8 + j;
      const int rg = wgBase + rt * 16 + m;
      if (rg < Btot) {
        const float v = acc2[nt][j];
        if (col < 60)       out_policy[(size_t)rg * 60 + col] = v + pb;
        else if (col == 60) out_value[rg] = tanhf(v + vbias);
      }
    }
  }
}

// ---------------------------------------------------------------------------
extern "C" void kernel_launch(void* const* d_in, const int* in_sizes, int n_in,
                              void* d_out, int out_size, void* d_ws, size_t ws_size,
                              hipStream_t stream) {
  const int*   stm_idx  = (const int*)d_in[0];
  const int*   nstm_idx = (const int*)d_in[1];
  const float* dense    = (const float*)d_in[2];
  const float* ft_w     = (const float*)d_in[3];
  const float* ft_b     = (const float*)d_in[4];
  const float* fc1_w    = (const float*)d_in[5];
  const float* fc1_b    = (const float*)d_in[6];
  const float* fc2v_w   = (const float*)d_in[7];
  const float* fc2v_b   = (const float*)d_in[8];
  const float* fc2p_w   = (const float*)d_in[9];
  const float* fc2p_b   = (const float*)d_in[10];

  const int Btot = in_sizes[0] / 3;
  float* out_policy = (float*)d_out;
  float* out_value  = out_policy + (size_t)Btot * 60;
  unsigned short* ws = (unsigned short*)d_ws;

  const int nconv = 256 * KP + N2 * HID;
  nnue_convert_weights<<<(nconv + 255) / 256, 256, 0, stream>>>(
      fc1_w, fc2p_w, fc2v_w, ws);

  const int nwg = (Btot + ROWS_PER_WG - 1) / ROWS_PER_WG;
  const size_t shmem =
      (size_t)ROWS_PER_WG * (ROWSTRIDE + HSTRIDE) * sizeof(unsigned short);
  nnue_fused_kernel<<<nwg, 256, shmem, stream>>>(
      stm_idx, nstm_idx, dense, ft_w, ft_b, fc1_b, fc2p_b, fc2v_b,
      (const unsigned short*)ws, out_policy, out_value, Btot);
}
